// CrossVarianceAttention_14431090114593
// MI455X (gfx1250) — matrix operand order repaired and checked by reading, not compiled, not run on hardware
//
#include <hip/hip_runtime.h>

// ---------------------------------------------------------------------------
// CrossVarianceAttention for MI455X (gfx1250, wave32, WMMA).
// All GEMM/attention math uses v_wmma_f32_16x16x32_bf16 (fp32 accumulate).
// Attention is fused per (b,h,16-q-row tile): score tile lives in LDS only.
// GEMM B-tiles and attention Q-tiles are staged to LDS with the CDNA5 async
// copy engine (global_load_async_to_lds_b128 + s_wait_asynccnt) when present.
// ---------------------------------------------------------------------------

#define DMODEL 512
#define NSEQ   1024
#define NB     8
#define NH     8
#define DK     64
#define MROWS  8192            // B*N
#define MD     ((size_t)MROWS * DMODEL)

typedef __attribute__((ext_vector_type(16))) __bf16 v16bf;
typedef __attribute__((ext_vector_type(8)))  float  v8f;
typedef int v4i_vs __attribute__((vector_size(16)));   // GCC-style V4i (builtin arg type)

union FragB16 {
    v16bf v;
    unsigned short u[16];
    uint4 q[2];
};

#if __has_builtin(__builtin_amdgcn_global_load_async_to_lds_b128)
#define ASYNC_LDS 1
#else
#define ASYNC_LDS 0
#endif

// 16-byte global -> LDS copy; async DMA path on CDNA5, sync fallback otherwise.
__device__ __forceinline__ void ld_async_b128(void* lds_dst, const void* g_src) {
#if ASYNC_LDS
    __builtin_amdgcn_global_load_async_to_lds_b128(
        (__attribute__((address_space(1))) v4i_vs*)g_src,
        (__attribute__((address_space(3))) v4i_vs*)lds_dst,
        0, 0);
#else
    *(uint4*)lds_dst = *(const uint4*)g_src;
#endif
}

// Wait until at most `pending` async-LDS copies remain in flight (this wave).
__device__ __forceinline__ void async_wait(int pending) {
#if ASYNC_LDS
#if __has_builtin(__builtin_amdgcn_s_wait_asynccnt)
    if (pending) __builtin_amdgcn_s_wait_asynccnt(1);
    else         __builtin_amdgcn_s_wait_asynccnt(0);
#else
    if (pending) asm volatile("s_wait_asynccnt 0x1" ::: "memory");
    else         asm volatile("s_wait_asynccnt 0x0" ::: "memory");
#endif
#else
    (void)pending;
#endif
}

__device__ inline unsigned short f2bf(float f) {
    union { float f; unsigned int u; } c;
    c.f = f;
    unsigned int u = c.u;
    unsigned int r = u + 0x7FFFu + ((u >> 16) & 1u);   // round-to-nearest-even
    return (unsigned short)(r >> 16);
}

__device__ inline v8f wmma_bf16(v16bf a, v16bf b, v8f c) {
    // D = A(16x32 bf16) * B(32x16 bf16) + C(16x16 f32)
    return __builtin_amdgcn_wmma_f32_16x16x32_bf16(
        /*neg_a=*/false, a, /*neg_b=*/false, b,
        /*c_mod=*/(short)0, c, /*reuse_a=*/false, /*reuse_b=*/false);
}

__device__ inline float wave_sum(float x) {
#pragma unroll
    for (int off = 16; off; off >>= 1) x += __shfl_xor(x, off, 32);
    return x;
}
__device__ inline float wave_max(float x) {
#pragma unroll
    for (int off = 16; off; off >>= 1) x = fmaxf(x, __shfl_xor(x, off, 32));
    return x;
}

// ---------------------------------------------------------------------------
// LayerNorm (one wave32 per 512-element row) -> bf16
// ---------------------------------------------------------------------------
__global__ __launch_bounds__(256) void k_layernorm_bf16(
    const float* __restrict__ x, const float* __restrict__ g,
    const float* __restrict__ b, unsigned short* __restrict__ y, int rows)
{
    int wv = threadIdx.x >> 5;
    int lane = threadIdx.x & 31;
    int row = blockIdx.x * 8 + wv;
    if (row >= rows) return;
    const float* xr = x + (size_t)row * DMODEL;
    float v[16];
    float s = 0.f, s2 = 0.f;
#pragma unroll
    for (int i = 0; i < 16; ++i) {
        v[i] = xr[lane + i * 32];
        s += v[i];
        s2 += v[i] * v[i];
    }
    s = wave_sum(s);
    s2 = wave_sum(s2);
    float mu = s * (1.f / DMODEL);
    float var = s2 * (1.f / DMODEL) - mu * mu;
    float rstd = rsqrtf(var + 1e-5f);
    unsigned short* yr = y + (size_t)row * DMODEL;
#pragma unroll
    for (int i = 0; i < 16; ++i) {
        int c = lane + i * 32;
        yr[c] = f2bf((v[i] - mu) * rstd * g[c] + b[c]);
    }
}

// fp32 -> bf16 elementwise
__global__ __launch_bounds__(256) void k_cvt_bf16(
    const float* __restrict__ x, unsigned short* __restrict__ y, int n)
{
    int i = blockIdx.x * 256 + threadIdx.x;
    if (i < n) y[i] = f2bf(x[i]);
}

// Wt[n][k] = bf16(W[k][n]) — LDS-tiled 512x512 transpose+convert
__global__ __launch_bounds__(256) void k_wt_bf16(
    const float* __restrict__ W, unsigned short* __restrict__ Wt)
{
    __shared__ float t[32][33];
    int bx = blockIdx.x & 15, by = blockIdx.x >> 4;
    int tx = threadIdx.x & 31, ty = threadIdx.x >> 5;
#pragma unroll
    for (int i = 0; i < 32; i += 8)
        t[ty + i][tx] = W[(size_t)(by * 32 + ty + i) * DMODEL + bx * 32 + tx];
    __syncthreads();
#pragma unroll
    for (int i = 0; i < 32; i += 8)
        Wt[(size_t)(bx * 32 + ty + i) * DMODEL + by * 32 + tx] = f2bf(t[tx][ty + i]);
}

// ---------------------------------------------------------------------------
// Generic bf16 WMMA GEMM:  C[M x 512] = A[M x 512] * Wt[512 x 512]^T + bias
// Wt stored [n][k] so B fragments (lane=column n, contiguous k) are b128 loads.
// Block tile 128(M) x 64(N); 8 waves, each 32x32 (2x2 WMMA 16x16 tiles).
// B tile (64x32 bf16, shared by all waves) is triple-buffered in LDS and
// streamed with the async copy engine: issue k+1, s_wait_asynccnt 1, barrier,
// compute k. Triple buffering makes one barrier per step WAR-safe.
// mode 0: fout row-major f32 [M,512]        (nontemporal store)
// mode 1: bout bf16 at [b,h,n,dk]           (Q/K layout)
// mode 2: bout bf16 at [b,h,dk,n]           (V transposed layout)
// ---------------------------------------------------------------------------
__global__ __launch_bounds__(256) void k_gemm_bf16(
    const unsigned short* __restrict__ A, const unsigned short* __restrict__ Wt,
    const float* __restrict__ bias, float* __restrict__ fout,
    unsigned short* __restrict__ bout, int mode)
{
    __shared__ unsigned short Bt[3][64][32];   // 12 KB, triple-buffered B tile

    const int tid = threadIdx.x;
    const int lane = tid & 31;
    const int wv = tid >> 5;
    const int wm = wv >> 1, wn = wv & 1;
    const int m0 = blockIdx.x * 128 + wm * 32;
    const int n0blk = blockIdx.y * 64;
    const int n0 = n0blk + wn * 32;
    const int lr = lane & 15;   // fragment row/col within tile
    const int lh = lane >> 4;   // lane half

    v8f acc[2][2] = {};
    const int aoff = lh * 8;    // A: K sub-offset per lane half
    const int boff = lh * 8;    // B: K sub-offset per lane half

    // stage a 64x32 B tile: 256 threads x 16B (row = tid>>2, 4 segs of 8 elems)
    const int srow = tid >> 2, sseg = tid & 3;
    auto stageB = [&](int buf, int kb) {
        ld_async_b128(&Bt[buf][srow][sseg * 8],
                      Wt + (size_t)(n0blk + srow) * DMODEL + kb + sseg * 8);
    };

    stageB(0, 0);
    for (int kc = 0; kc < 16; ++kc) {
        const int kb = kc * 32;
        const bool hasNext = (kc + 1) < 16;
        if (hasNext) stageB((kc + 1) % 3, kb + 32);
        async_wait(hasNext ? 1 : 0);   // stage kc retired, kc+1 still in flight
        __syncthreads();

        const int bufr = kc % 3;
        FragB16 a[2], b[2];
#pragma unroll
        for (int i = 0; i < 2; ++i) {
            const unsigned short* ap =
                A + (size_t)(m0 + i * 16 + lr) * DMODEL + kb + aoff;
            a[i].q[0] = *(const uint4*)(ap);
            a[i].q[1] = *(const uint4*)(ap + 16);
        }
#pragma unroll
        for (int j = 0; j < 2; ++j) {
            const unsigned short* bp = &Bt[bufr][wn * 32 + j * 16 + lr][boff];
            b[j].q[0] = *(const uint4*)(bp);
            b[j].q[1] = *(const uint4*)(bp + 16);
        }
        if (hasNext)   // pull the next A cachelines (global_prefetch_b8)
            __builtin_prefetch(A + (size_t)(m0 + lr) * DMODEL + kb + 64, 0, 1);
#pragma unroll
        for (int i = 0; i < 2; ++i)
#pragma unroll
            for (int j = 0; j < 2; ++j)
                acc[i][j] = wmma_bf16(a[i].v, b[j].v, acc[i][j]);
    }

#pragma unroll
    for (int i = 0; i < 2; ++i)
#pragma unroll
        for (int j = 0; j < 2; ++j)
#pragma unroll
            for (int r = 0; r < 8; ++r) {
                int m = m0 + i * 16 + r + lh * 8;
                int c = n0 + j * 16 + lr;
                float val = acc[i][j][r] + bias[c];
                if (mode == 0) {
                    __builtin_nontemporal_store(val, &fout[(size_t)m * DMODEL + c]);
                } else {
                    int bb = m >> 10, n = m & 1023, h = c >> 6, dk = c & 63;
                    size_t idx = (mode == 1)
                        ? (((size_t)(bb * NH + h) * NSEQ + n) * DK + dk)
                        : (((size_t)(bb * NH + h) * DK + dk) * NSEQ + n);
                    bout[idx] = f2bf(val);
                }
            }
}

// ---------------------------------------------------------------------------
// Fused attention per (bh, 16-row q tile):
//   S = Q K^T / 8            (WMMA, f32 in LDS)
//   row mean/var -> sigmoid variance weighting -> softmax (max-subtracted)
//   O = softmax(S) V         (WMMA, split-K over wave pairs)
// Q:[bh,n,dk] bf16, K:[bh,n,dk] bf16, Vt:[bh,dk,n] bf16, O:[8192,512] bf16
// LDS: 64K(S) + 32K(P) + 8K(Opart) + 2K(Qt) ≈ 106 KB  (< 160 KB CU-mode)
// ---------------------------------------------------------------------------
__global__ __launch_bounds__(256) void k_attn(
    const unsigned short* __restrict__ Q, const unsigned short* __restrict__ K,
    const unsigned short* __restrict__ Vt, unsigned short* __restrict__ O)
{
    __shared__ float S[16][1024];
    __shared__ unsigned short P[16][1024];
    __shared__ float Opart[2][16][64];
    __shared__ float rinv[16];
    __shared__ unsigned short Qt[16][64];

    const int tid = threadIdx.x;
    const int lane = tid & 31, wv = tid >> 5;
    const int lr = lane & 15, lh = lane >> 4;
    const int bh = blockIdx.y;
    const int qb = blockIdx.x * 16;

    // stage Q tile (16x64 bf16 = 2 KB) into LDS via the async engine
    if (tid < 128) {
        int row = tid >> 3, seg = tid & 7;
        ld_async_b128(&Qt[row][seg * 8],
                      Q + ((size_t)bh * NSEQ + qb + row) * DK + seg * 8);
    }
    async_wait(0);
    __syncthreads();

    // A fragments of Q for both d-chunks (d = 0..31, 32..63), reused 8x per wave
    FragB16 aq[2];
#pragma unroll
    for (int ch = 0; ch < 2; ++ch) {
        aq[ch].q[0] = *(const uint4*)&Qt[lr][ch * 32 + lh * 8];
        aq[ch].q[1] = *(const uint4*)&Qt[lr][ch * 32 + lh * 8 + 16];
    }

    // ---- S = Q K^T / sqrt(dk): 64 key tiles of 16, 8 per wave --------------
    for (int t = 0; t < 8; ++t) {
        int kc = (wv + t * 8) * 16;   // key base
        v8f acc = {};
#pragma unroll
        for (int ch = 0; ch < 2; ++ch) {
            FragB16 bk;   // B = K^T chunk: lane -> key column, contiguous d
            const unsigned short* kp =
                K + ((size_t)bh * NSEQ + kc + lr) * DK + ch * 32 + lh * 8;
            bk.q[0] = *(const uint4*)kp;
            bk.q[1] = *(const uint4*)(kp + 16);
            acc = wmma_bf16(aq[ch].v, bk.v, acc);
        }
#pragma unroll
        for (int r = 0; r < 8; ++r)
            S[r + lh * 8][kc + lr] = acc[r] * 0.125f;
    }
    __syncthreads();

    // ---- variance weighting + softmax, one wave per 2 rows -----------------
    for (int rr = wv; rr < 16; rr += 8) {
        float s = 0.f, s2 = 0.f;
        for (int i = 0; i < 32; ++i) {
            float x = S[rr][lane + i * 32];
            s += x;
            s2 += x * x;
        }
        s = wave_sum(s);
        s2 = wave_sum(s2);
        float mean = s * (1.f / 1024.f);
        float var = s2 * (1.f / 1024.f) - mean * mean;
        float wden = 1.f / (2.f * var + 1e-6f);
        float mx = -3.0e38f;
        for (int i = 0; i < 32; ++i) {
            int c = lane + i * 32;
            float x = S[rr][c];
            float cs = (x - mean) * (x - mean);
            float wgt = 1.f / (1.f + __expf(-(cs * wden + 0.5f)));
            float y = x * wgt;
            S[rr][c] = y;            // same lane/slot: race-free overwrite
            mx = fmaxf(mx, y);
        }
        mx = wave_max(mx);
        float se = 0.f;
        for (int i = 0; i < 32; ++i) {
            int c = lane + i * 32;
            float p = __expf(S[rr][c] - mx);
            se += p;
            P[rr][c] = f2bf(p);
        }
        se = wave_sum(se);
        if (lane == 0) rinv[rr] = 1.f / se;
    }
    __syncthreads();

    // ---- O = P V: 4 d-tiles x 2 k-halves across 8 waves --------------------
    {
        int dt = wv & 3, kh = wv >> 2;
        v8f acc = {};
        for (int c = 0; c < 16; ++c) {
            int kb2 = kh * 512 + c * 32;
            FragB16 ap, bv;
            ap.q[0] = *(const uint4*)&P[lr][kb2 + lh * 8];
            ap.q[1] = *(const uint4*)&P[lr][kb2 + lh * 8 + 16];
            const unsigned short* vp =   // B = V chunk: lane -> d column, contiguous k
                Vt + ((size_t)bh * DK + dt * 16 + lr) * NSEQ + kb2 + lh * 8;
            bv.q[0] = *(const uint4*)vp;
            bv.q[1] = *(const uint4*)(vp + 16);
            acc = wmma_bf16(ap.v, bv.v, acc);
        }
#pragma unroll
        for (int r = 0; r < 8; ++r)
            Opart[kh][r + lh * 8][dt * 16 + lr] = acc[r];
    }
    __syncthreads();

    // ---- combine split-K, normalize, scatter to merged layout --------------
    int b = bh >> 3, h = bh & 7;
#pragma unroll
    for (int it = 0; it < 4; ++it) {
        int idx = tid + it * 256;
        int row = idx >> 6, col = idx & 63;
        float v = (Opart[0][row][col] + Opart[1][row][col]) * rinv[row];
        O[((size_t)b * NSEQ + qb + row) * DMODEL + h * DK + col] = f2bf(v);
    }
}

// ---------------------------------------------------------------------------
// Host-side orchestration
// ---------------------------------------------------------------------------
extern "C" void kernel_launch(void* const* d_in, const int* in_sizes, int n_in,
                              void* d_out, int out_size, void* d_ws, size_t ws_size,
                              hipStream_t stream)
{
    (void)in_sizes; (void)n_in; (void)out_size; (void)ws_size;

    const float* rgb_fea   = (const float*)d_in[0];
    const float* ir_fea    = (const float*)d_in[1];
    const float* rgb_fused = (const float*)d_in[2];
    const float* ir_fused  = (const float*)d_in[3];
    const float* Wq_vis = (const float*)d_in[4];  const float* bq_vis = (const float*)d_in[5];
    const float* Wk_vis = (const float*)d_in[6];  const float* bk_vis = (const float*)d_in[7];
    const float* Wv_vis = (const float*)d_in[8];  const float* bv_vis = (const float*)d_in[9];
    const float* Wq_ir  = (const float*)d_in[10]; const float* bq_ir  = (const float*)d_in[11];
    const float* Wk_ir  = (const float*)d_in[12]; const float* bk_ir  = (const float*)d_in[13];
    const float* Wv_ir  = (const float*)d_in[14]; const float* bv_ir  = (const float*)d_in[15];
    const float* Wo_vis = (const float*)d_in[16]; const float* bo_vis = (const float*)d_in[17];
    const float* Wo_ir  = (const float*)d_in[18]; const float* bo_ir  = (const float*)d_in[19];
    const float* ln1_g = (const float*)d_in[20];  const float* ln1_b = (const float*)d_in[21];
    const float* ln2_g = (const float*)d_in[22];  const float* ln2_b = (const float*)d_in[23];

    char* wp = (char*)d_ws;
    auto take = [&](size_t bytes) -> void* {
        void* p = wp;
        wp += (bytes + 255) & ~(size_t)255;
        return p;
    };
    typedef unsigned short us;
    us* rgbN  = (us*)take(MD * 2);   // LN(rgb_fea) bf16
    us* irN   = (us*)take(MD * 2);   // LN(ir_fea)  bf16
    us* rgbF  = (us*)take(MD * 2);   // rgb_fused bf16
    us* irF   = (us*)take(MD * 2);   // ir_fused  bf16
    us* WtQv  = (us*)take(DMODEL * DMODEL * 2);
    us* WtKv  = (us*)take(DMODEL * DMODEL * 2);
    us* WtVv  = (us*)take(DMODEL * DMODEL * 2);
    us* WtQi  = (us*)take(DMODEL * DMODEL * 2);
    us* WtKi  = (us*)take(DMODEL * DMODEL * 2);
    us* WtVi  = (us*)take(DMODEL * DMODEL * 2);
    us* WtOv  = (us*)take(DMODEL * DMODEL * 2);
    us* WtOi  = (us*)take(DMODEL * DMODEL * 2);
    us* q_vis = (us*)take(MD * 2);
    us* k_vis = (us*)take(MD * 2);
    us* v_vis = (us*)take(MD * 2);   // transposed [bh,dk,n]
    us* q_ir  = (us*)take(MD * 2);
    us* k_ir  = (us*)take(MD * 2);
    us* v_ir  = (us*)take(MD * 2);   // transposed
    us* Ovis  = (us*)take(MD * 2);
    us* Oir   = (us*)take(MD * 2);

    // 1) LayerNorm -> bf16, fused inputs -> bf16, weights -> bf16 transposed
    k_layernorm_bf16<<<MROWS / 8, 256, 0, stream>>>(rgb_fea, ln1_g, ln1_b, rgbN, MROWS);
    k_layernorm_bf16<<<MROWS / 8, 256, 0, stream>>>(ir_fea,  ln2_g, ln2_b, irN,  MROWS);
    k_cvt_bf16<<<(int)(MD / 256), 256, 0, stream>>>(rgb_fused, rgbF, (int)MD);
    k_cvt_bf16<<<(int)(MD / 256), 256, 0, stream>>>(ir_fused,  irF,  (int)MD);
    k_wt_bf16<<<256, 256, 0, stream>>>(Wq_vis, WtQv);
    k_wt_bf16<<<256, 256, 0, stream>>>(Wk_vis, WtKv);
    k_wt_bf16<<<256, 256, 0, stream>>>(Wv_vis, WtVv);
    k_wt_bf16<<<256, 256, 0, stream>>>(Wq_ir,  WtQi);
    k_wt_bf16<<<256, 256, 0, stream>>>(Wk_ir,  WtKi);
    k_wt_bf16<<<256, 256, 0, stream>>>(Wv_ir,  WtVi);
    k_wt_bf16<<<256, 256, 0, stream>>>(Wo_vis, WtOv);
    k_wt_bf16<<<256, 256, 0, stream>>>(Wo_ir,  WtOi);

    // 2) QKV projections (WMMA GEMM, head-layout epilogues)
    dim3 gg(MROWS / 128, DMODEL / 64);
    k_gemm_bf16<<<gg, 256, 0, stream>>>(rgbF, WtQv, bq_vis, nullptr, q_vis, 1);
    k_gemm_bf16<<<gg, 256, 0, stream>>>(rgbN, WtKv, bk_vis, nullptr, k_vis, 1);
    k_gemm_bf16<<<gg, 256, 0, stream>>>(rgbN, WtVv, bv_vis, nullptr, v_vis, 2);
    k_gemm_bf16<<<gg, 256, 0, stream>>>(irF,  WtQi, bq_ir,  nullptr, q_ir,  1);
    k_gemm_bf16<<<gg, 256, 0, stream>>>(irN,  WtKi, bk_ir,  nullptr, k_ir,  1);
    k_gemm_bf16<<<gg, 256, 0, stream>>>(irN,  WtVi, bv_ir,  nullptr, v_ir,  2);

    // 3) fused cross attention: att_vis(q_ir, k_vis, v_vis), att_ir(q_vis, k_ir, v_ir)
    dim3 ga(NSEQ / 16, NB * NH);
    k_attn<<<ga, 256, 0, stream>>>(q_ir,  k_vis, v_vis, Ovis);
    k_attn<<<ga, 256, 0, stream>>>(q_vis, k_ir,  v_ir,  Oir);

    // 4) output projections -> d_out (out_vis ++ out_ir, f32)
    float* out = (float*)d_out;
    k_gemm_bf16<<<gg, 256, 0, stream>>>(Ovis, WtOv, bo_vis, out,      nullptr, 0);
    k_gemm_bf16<<<gg, 256, 0, stream>>>(Oir,  WtOi, bo_ir,  out + MD, nullptr, 0);
}
